// ChannelAttentionBlock3d_78134045049361
// MI455X (gfx1250) — compile-verified, matrix operand order checked by reading, not run on hardware
//
#include <hip/hip_runtime.h>

#define B_   4
#define C_   128
#define N_   4096
#define G_   8
#define CPG  16
#define O3C  384   // 3*C

typedef __attribute__((ext_vector_type(16))) _Float16 v16h;
typedef __attribute__((ext_vector_type(8)))  _Float16 v8h;
typedef __attribute__((ext_vector_type(8)))  float    v8f;

// ---------------------------------------------------------------------------
// Per-lane operand fetch for v_wmma_f32_16x16x32_f16.
// A-operand (16xK=32, 16-bit): lane L holds row M=L&15; K elements
//   {0..7,16..23} for L<16 and {8..15,24..31} for L>=16 (ISA 7.12.2).
// B-operand uses the mirrored pattern (N striped across lanes), so both
// operands fetch identically from a row-major [row][K] f16 buffer:
// two 16-byte loads per operand (global_load_b128 or ds_load_b128).
// ---------------------------------------------------------------------------
__device__ __forceinline__ v16h load_ab(const _Float16* __restrict__ row,
                                        int kbase, int laneHi) {
  const int lo = kbase + (laneHi ? 8 : 0);
  v8h a = *(const v8h*)(row + lo);
  v8h b = *(const v8h*)(row + lo + 16);
  v16h r;
#pragma unroll
  for (int i = 0; i < 8; ++i) { r[i] = a[i]; r[i + 8] = b[i]; }
  return r;
}

__device__ __forceinline__ v8f wmma16(v16h a, v16h b, v8f c) {
  return __builtin_amdgcn_wmma_f32_16x16x32_f16(false, a, false, b,
                                                (short)0, c, false, false);
}

// CDNA5 async global->LDS copy (tracked by ASYNCcnt, ISA 10 / 15.18.3).
__device__ __forceinline__ void async_copy_b128(unsigned lds_off,
                                                const void* gaddr) {
  asm volatile("global_load_async_to_lds_b128 %0, %1, off"
               :: "v"(lds_off), "v"(gaddr) : "memory");
}
__device__ __forceinline__ void wait_async0() {
  asm volatile("s_wait_asynccnt 0x0" ::: "memory");
}

// ---------------------------------------------------------------------------
// K1: GroupNorm statistics. One block per (b, group): reduce 16ch x 4096.
// Groups are contiguous 16-channel blocks, so base = bg * 16 * N.
// ---------------------------------------------------------------------------
__global__ __launch_bounds__(256) void gn_stats(const float* __restrict__ x,
                                                float* __restrict__ stats) {
  const int bg = blockIdx.x;                     // 0..31
  const float* p = x + (size_t)bg * CPG * N_;
  float s = 0.f, s2 = 0.f;
  for (int i = threadIdx.x; i < CPG * N_; i += 256) {
    float v = p[i];
    s += v; s2 += v * v;
  }
  __shared__ float rs[256], rq[256];
  rs[threadIdx.x] = s; rq[threadIdx.x] = s2;
  __syncthreads();
  for (int off = 128; off > 0; off >>= 1) {
    if (threadIdx.x < off) {
      rs[threadIdx.x] += rs[threadIdx.x + off];
      rq[threadIdx.x] += rq[threadIdx.x + off];
    }
    __syncthreads();
  }
  if (threadIdx.x == 0) {
    const float invn = 1.0f / (float)(CPG * N_);
    float mean = rs[0] * invn;
    float var  = rq[0] * invn - mean * mean;
    stats[bg * 2 + 0] = mean;
    stats[bg * 2 + 1] = rsqrtf(var + 1e-6f);
  }
}

// ---------------------------------------------------------------------------
// K2: f32 -> f16 weight conversion (w_qkv and w_proj, row-major kept).
// ---------------------------------------------------------------------------
__global__ __launch_bounds__(256) void cvt_w(const float* __restrict__ wqkv,
                                             const float* __restrict__ wproj,
                                             _Float16* __restrict__ w16,
                                             _Float16* __restrict__ wp16) {
  int i = blockIdx.x * 256 + threadIdx.x;
  if (i < O3C * C_) w16[i]  = (_Float16)wqkv[i];
  if (i < C_ * C_)  wp16[i] = (_Float16)wproj[i];
}

// ---------------------------------------------------------------------------
// K3: apply GroupNorm and transpose x[B][C][N] -> xnT[B][N][C] (f16) so the
// GEMM B-operand reads contiguous K(=C) runs. 32x32 LDS tile transpose.
// grid (N/32, C/32, B), block (32,8).
// ---------------------------------------------------------------------------
__global__ __launch_bounds__(256) void norm_transpose(
    const float* __restrict__ x, const float* __restrict__ stats,
    const float* __restrict__ gamma, const float* __restrict__ beta,
    _Float16* __restrict__ xnT) {
  __shared__ _Float16 tile[32][33];
  const int b = blockIdx.z, c0 = blockIdx.y * 32, n0 = blockIdx.x * 32;
  const int tx = threadIdx.x, ty = threadIdx.y;
#pragma unroll
  for (int i = 0; i < 4; ++i) {
    const int cl = ty + i * 8;
    const int c  = c0 + cl;
    const int g  = c >> 4;
    const float mean = stats[(b * G_ + g) * 2 + 0];
    const float rstd = stats[(b * G_ + g) * 2 + 1];
    const float v = x[((size_t)b * C_ + c) * N_ + n0 + tx];
    tile[cl][tx] = (_Float16)(((v - mean) * rstd) * gamma[c] + beta[c]);
  }
  __syncthreads();
#pragma unroll
  for (int i = 0; i < 4; ++i) {
    const int nl = ty + i * 8;
    xnT[((size_t)b * N_ + n0 + nl) * C_ + c0 + tx] = tile[tx][nl];
  }
}

// ---------------------------------------------------------------------------
// K4: QKV GEMM. One wave per 16x16 output tile, K-loop over C=128 (4 WMMAs).
// q,k written transposed ([B][N][C], f16, q pre-scaled by C^-0.5);
// v written [B][C][N] (f16) for the PV A-operand.
// waves = B * (N/16) * (3C/16) = 24576; block = 256 (8 waves).
// ---------------------------------------------------------------------------
__global__ __launch_bounds__(256) void qkv_gemm(
    const _Float16* __restrict__ w16, const _Float16* __restrict__ xnT,
    const float* __restrict__ bqkv,
    _Float16* __restrict__ qT, _Float16* __restrict__ kT,
    _Float16* __restrict__ v16) {
  const int wave = (blockIdx.x * 256 + threadIdx.x) >> 5;
  const int lane = threadIdx.x & 31;
  const int ot = wave % 24;
  const int nt = (wave / 24) % 256;
  const int b  = wave / (24 * 256);
  const int o0 = ot * 16, n0 = nt * 16;
  const int laneHi = lane >> 4, lr = lane & 15;

  const _Float16* arow = w16 + (size_t)(o0 + lr) * C_;
  const _Float16* brow = xnT + ((size_t)b * N_ + n0 + lr) * C_;

  v8f acc = {};
#pragma unroll
  for (int kc = 0; kc < C_; kc += 32)
    acc = wmma16(load_ab(arow, kc, laneHi), load_ab(brow, kc, laneHi), acc);

  // D layout: lane holds column n=n0+lr, rows o = o0 + laneHi*8 + r.
  const int obase = o0 + laneHi * 8;
  const int n = n0 + lr;
  if (o0 < C_) {                      // q: scaled, transposed store
    const float scale = 0.08838834764831845f;  // 128^-0.5
    v8h outv;
#pragma unroll
    for (int r = 0; r < 8; ++r)
      outv[r] = (_Float16)((acc[r] + bqkv[obase + r]) * scale);
    *(v8h*)(qT + ((size_t)b * N_ + n) * C_ + obase) = outv;
  } else if (o0 < 2 * C_) {           // k: transposed store
    v8h outv;
#pragma unroll
    for (int r = 0; r < 8; ++r)
      outv[r] = (_Float16)(acc[r] + bqkv[obase + r]);
    *(v8h*)(kT + ((size_t)b * N_ + n) * C_ + (obase - C_)) = outv;
  } else {                            // v: [C][N] scatter
#pragma unroll
    for (int r = 0; r < 8; ++r)
      v16[((size_t)b * C_ + (obase - 2 * C_) + r) * N_ + n] =
          (_Float16)(acc[r] + bqkv[obase + r]);
  }
}

// ---------------------------------------------------------------------------
// K5: flash attention over channels-contraction.
// Block = 8 waves, all on the SAME batch and 8 consecutive 16-query tiles,
// so the 32-key K/V chunk is shared: staged into LDS once per block with
// async global->LDS b128 copies (ASYNCcnt), double-buffered, DMA of chunk
// i+1 overlapped with WMMAs on chunk i. Operands then come from ds_load.
// Per 32-key chunk per wave: St = Kt x Q (8 WMMAs), online softmax (in-lane
// + one shfl_xor(16)), in-lane D->B conversion of P, O^T += V x P^T (8 WMMAs).
// grid = B * (N/16) / 8 = 128 blocks.
// ---------------------------------------------------------------------------
__global__ __launch_bounds__(256) void attn(
    const _Float16* __restrict__ qT, const _Float16* __restrict__ kT,
    const _Float16* __restrict__ v16, _Float16* __restrict__ hT) {
  __shared__ alignas(16) _Float16 Kb[2][32][128];  // 16 KB  [key][c]
  __shared__ alignas(16) _Float16 Vb[2][128][32];  // 16 KB  [c][key]

  const int tid  = threadIdx.x;
  const int lane = tid & 31;
  const int w    = tid >> 5;                 // wave in block: q-subtile
  const int b    = blockIdx.x >> 5;          // 32 blocks per batch
  const int q0   = ((blockIdx.x & 31) * 8 + w) * 16;
  const int laneHi = lane >> 4, lr = lane & 15;

  const _Float16* kbase = kT + (size_t)b * N_ * C_;
  const _Float16* vbase = v16 + (size_t)b * C_ * N_;

  // Q is the B-operand of St = Kt x Q: preload the 4 K-steps over C=128.
  const _Float16* qrow = qT + ((size_t)b * N_ + q0 + lr) * C_;
  v16h Qb[4];
#pragma unroll
  for (int s = 0; s < 4; ++s) Qb[s] = load_ab(qrow, s * 32, laneHi);

  v8f O[8];
  v8f z = {};
#pragma unroll
  for (int t = 0; t < 8; ++t) O[t] = z;
  float m = -3.0e38f, l = 0.f;

  // ---- async stage of chunk kc into buffer buf (all 256 threads) ----------
  // K tile: 32 rows x 128 halves = one flat 8 KB region (rows contiguous
  //         in kT[B][N][C]); thread t copies 16B at t*16 and t*16+4096.
  // V tile: 128 rows x 32 halves (64B rows, global stride N*2); 512 b128
  //         transfers, thread t does i = t and i = t+256: c=i>>2, seg=i&3.
  auto stage = [&](int buf, int kc) {
    const char* ksrc = (const char*)(kbase + (size_t)kc * C_);
    unsigned kdst = (unsigned)(size_t)&Kb[buf][0][0];
    async_copy_b128(kdst + tid * 16,        ksrc + tid * 16);
    async_copy_b128(kdst + tid * 16 + 4096, ksrc + tid * 16 + 4096);
    unsigned vdst = (unsigned)(size_t)&Vb[buf][0][0];
#pragma unroll
    for (int p = 0; p < 2; ++p) {
      const int i = tid + p * 256;
      const int c = i >> 2, seg = i & 3;
      async_copy_b128(vdst + (unsigned)(c * 64 + seg * 16),
                      (const char*)(vbase + (size_t)c * N_ + kc + seg * 8));
    }
  };

  stage(0, 0);

  for (int kc = 0; kc < N_; kc += 32) {
    const int buf = (kc >> 5) & 1;
    wait_async0();            // chunk kc resident in LDS (this wave's DMA)
    __syncthreads();          // ... and everyone else's
    if (kc + 32 < N_) stage(buf ^ 1, kc + 32);   // overlap next DMA

    const _Float16* krow0 = &Kb[buf][lr][0];
    const _Float16* krow1 = &Kb[buf][16 + lr][0];

    v8f st0 = z, st1 = z;
#pragma unroll
    for (int s = 0; s < 4; ++s)
      st0 = wmma16(load_ab(krow0, s * 32, laneHi), Qb[s], st0);
#pragma unroll
    for (int s = 0; s < 4; ++s)
      st1 = wmma16(load_ab(krow1, s * 32, laneHi), Qb[s], st1);

    // Lane holds, for query q=q0+lr, keys {laneHi*8 + r} (st0) and
    // {16 + laneHi*8 + r} (st1); partner lane (lane^16) holds the rest.
    float cmax = st0[0];
#pragma unroll
    for (int r = 1; r < 8; ++r) cmax = fmaxf(cmax, st0[r]);
#pragma unroll
    for (int r = 0; r < 8; ++r) cmax = fmaxf(cmax, st1[r]);
    cmax = fmaxf(cmax, __shfl_xor(cmax, 16, 32));

    const float newm  = fmaxf(m, cmax);
    const float alpha = __expf(m - newm);
    float psum = 0.f;
    v16h pB;
#pragma unroll
    for (int r = 0; r < 8; ++r) {
      float e = __expf(st0[r] - newm);
      psum += e; pB[r] = (_Float16)e;
    }
#pragma unroll
    for (int r = 0; r < 8; ++r) {
      float e = __expf(st1[r] - newm);
      psum += e; pB[8 + r] = (_Float16)e;
    }
    psum += __shfl_xor(psum, 16, 32);
    l = l * alpha + psum;
    m = newm;

#pragma unroll
    for (int t = 0; t < 8; ++t)
#pragma unroll
      for (int r = 0; r < 8; ++r) O[t][r] *= alpha;

    // O^T += V x P^T, tiled over C in 8 steps of 16 channels (from LDS).
#pragma unroll
    for (int t = 0; t < 8; ++t)
      O[t] = wmma16(load_ab(&Vb[buf][t * 16 + lr][0], 0, laneHi), pB, O[t]);

    __syncthreads();          // protect buf before iteration kc+64 rewrites it
  }

  const float inv = 1.0f / l;
  const int qn = q0 + lr;
#pragma unroll
  for (int t = 0; t < 8; ++t) {
    v8h outv;
#pragma unroll
    for (int r = 0; r < 8; ++r) outv[r] = (_Float16)(O[t][r] * inv);
    *(v8h*)(hT + ((size_t)b * N_ + qn) * C_ + t * 16 + laneHi * 8) = outv;
  }
}

// ---------------------------------------------------------------------------
// K6: proj GEMM + bias + residual. One wave per 16x16 tile of out[B][C][N].
// waves = B * (N/16) * (C/16) = 8192; block = 256.
// ---------------------------------------------------------------------------
__global__ __launch_bounds__(256) void proj_skip(
    const _Float16* __restrict__ wp16, const _Float16* __restrict__ hT,
    const float* __restrict__ bproj, const float* __restrict__ x,
    float* __restrict__ out) {
  const int wave = (blockIdx.x * 256 + threadIdx.x) >> 5;
  const int lane = threadIdx.x & 31;
  const int ot = wave % 8;
  const int nt = (wave / 8) % 256;
  const int b  = wave / (8 * 256);
  const int o0 = ot * 16, n0 = nt * 16;
  const int laneHi = lane >> 4, lr = lane & 15;

  const _Float16* arow = wp16 + (size_t)(o0 + lr) * C_;
  const _Float16* brow = hT + ((size_t)b * N_ + n0 + lr) * C_;

  v8f acc = {};
#pragma unroll
  for (int kc = 0; kc < C_; kc += 32)
    acc = wmma16(load_ab(arow, kc, laneHi), load_ab(brow, kc, laneHi), acc);

  const int obase = o0 + laneHi * 8;
  const int n = n0 + lr;
#pragma unroll
  for (int r = 0; r < 8; ++r) {
    const size_t idx = ((size_t)b * C_ + obase + r) * N_ + n;
    out[idx] = x[idx] + bproj[obase + r] + acc[r];
  }
}

// ---------------------------------------------------------------------------
// Workspace layout (bytes, all offsets 1 KB aligned). Total ~21.1 MB.
// ---------------------------------------------------------------------------
#define WS_STATS 0u
#define WS_W16   1024u
#define WS_WP16  (WS_W16 + (size_t)O3C * C_ * 2)            //  99328
#define WS_XNT   (WS_WP16 + (size_t)C_ * C_ * 2)            // 132096
#define WS_QT    (WS_XNT + (size_t)B_ * N_ * C_ * 2)
#define WS_KT    (WS_QT + (size_t)B_ * N_ * C_ * 2)
#define WS_V     (WS_KT + (size_t)B_ * N_ * C_ * 2)
#define WS_HT    (WS_V + (size_t)B_ * N_ * C_ * 2)

extern "C" void kernel_launch(void* const* d_in, const int* in_sizes, int n_in,
                              void* d_out, int out_size, void* d_ws, size_t ws_size,
                              hipStream_t stream) {
  (void)in_sizes; (void)n_in; (void)out_size; (void)ws_size;
  const float* x     = (const float*)d_in[0];
  const float* gamma = (const float*)d_in[1];
  const float* beta  = (const float*)d_in[2];
  const float* wqkv  = (const float*)d_in[3];
  const float* bqkv  = (const float*)d_in[4];
  const float* wproj = (const float*)d_in[5];
  const float* bproj = (const float*)d_in[6];
  float* out = (float*)d_out;

  char* ws = (char*)d_ws;
  float*    stats = (float*)(ws + WS_STATS);
  _Float16* w16   = (_Float16*)(ws + WS_W16);
  _Float16* wp16  = (_Float16*)(ws + WS_WP16);
  _Float16* xnT   = (_Float16*)(ws + WS_XNT);
  _Float16* qT    = (_Float16*)(ws + WS_QT);
  _Float16* kT    = (_Float16*)(ws + WS_KT);
  _Float16* v16   = (_Float16*)(ws + WS_V);
  _Float16* hT    = (_Float16*)(ws + WS_HT);

  gn_stats<<<B_ * G_, 256, 0, stream>>>(x, stats);
  cvt_w<<<(O3C * C_ + 255) / 256, 256, 0, stream>>>(wqkv, wproj, w16, wp16);
  norm_transpose<<<dim3(N_ / 32, C_ / 32, B_), dim3(32, 8), 0, stream>>>(
      x, stats, gamma, beta, xnT);
  qkv_gemm<<<(B_ * (N_ / 16) * (O3C / 16)) / 8, 256, 0, stream>>>(
      w16, xnT, bqkv, qT, kT, v16);
  attn<<<(B_ * (N_ / 16)) / 8, 256, 0, stream>>>(qT, kT, v16, hT);
  proj_skip<<<(B_ * (N_ / 16) * (C_ / 16)) / 8, 256, 0, stream>>>(
      wp16, hT, bproj, x, out);
}
